// WaveFieldAttention_31207232373164
// MI455X (gfx1250) — compile-verified, matrix operand order checked by reading, not run on hardware
//
#include <hip/hip_runtime.h>
#include <math.h>

// ---------------- problem constants (match reference) ----------------
#define DIMC   1024
#define HC     16
#define DHC    64
#define GC     512
#define BC     4
#define NTOK   4096
#define MTOK   (BC * NTOK)          // 16384 rows for token GEMMs
#define STRIDE_F (511.0f / 4095.0f) // (G-1)/(MAX_SEQ-1)
#define SCALE_F  8.0f               // sqrt(DH)
#define TWO_PI_F 6.28318530717958647692f

typedef __attribute__((ext_vector_type(16))) __bf16 v16bf;
typedef __attribute__((ext_vector_type(8)))  __bf16 v8bf;
typedef __attribute__((ext_vector_type(8)))  float  v8f;
typedef __attribute__((ext_vector_type(4)))  float  v4f;

static __device__ __forceinline__ unsigned short f2bf(float f) {
  unsigned int u = __builtin_bit_cast(unsigned int, f);
  u = u + 0x7FFFu + ((u >> 16) & 1u);           // round-to-nearest-even
  return (unsigned short)(u >> 16);
}
static __device__ __forceinline__ __bf16 us2bf(unsigned short s) {
  return __builtin_bit_cast(__bf16, s);
}

#define CAT16(lo, hi) __builtin_shufflevector(lo, hi, 0,1,2,3,4,5,6,7,8,9,10,11,12,13,14,15)

// ---------------------------------------------------------------------
// Tiled GEMM, C[M,N] = A[M,K] @ B[K,N] (+ bias[N]), f32 in/out, bf16 WMMA.
// 128 threads = 4 waves; block tile 64x64; K step 32.
// REQUIRES: M%64==0, N%64==0, K%32==0 (true for all call sites).
// LDS tiles are K-contiguous per lane-fragment => ds_load_b128 fragment loads.
// Batched via blockIdx.z: A offset (z % aMod)*aStride, B/C offset z*stride.
// ---------------------------------------------------------------------
__global__ void __launch_bounds__(128)
gemm_bf16_wmma(const float* __restrict__ A, const float* __restrict__ Bw,
               const float* __restrict__ bias, float* __restrict__ C,
               int M, int N, int K, int ldA, int ldB, int ldC,
               long aStride, int aMod, long bStride, long cStride)
{
  __shared__ __align__(16) __bf16 As[64][40];   // [row][k], 80B row stride (16B mult)
  __shared__ __align__(16) __bf16 Bst[64][40];  // [col][k], transposed B tile

  const int tid  = threadIdx.x;
  const int lane = tid & 31;
  const int wave = tid >> 5;
  const int half = lane >> 4;     // K-subblock selector (ISA 16-bit layout)
  const int l16  = lane & 15;

  const int rowBase = blockIdx.y * 64;
  const int colBase = blockIdx.x * 64;
  const int z = blockIdx.z;

  const float* Ap = A  + (long)(z % aMod) * aStride;
  const float* Bp = Bw + (long)z * bStride;
  float*       Cp = C  + (long)z * cStride;

  v8f acc[4];
  acc[0] = (v8f){}; acc[1] = (v8f){}; acc[2] = (v8f){}; acc[3] = (v8f){};

  const int klo = 8 * half;
  const int khi = 16 + 8 * half;
  const int am  = wave * 16 + l16;

  for (int kb = 0; kb < K; kb += 32) {
    // ---- A tile 64x32: 256 chunks of 8 floats, 2 per thread, packed b128 store
    #pragma unroll
    for (int s = 0; s < 2; ++s) {
      int task = tid + s * 128;          // 0..255
      int r  = task >> 2;                // 0..63
      int c0 = (task & 3) * 8;           // 0,8,16,24
      const float* src = Ap + (long)(rowBase + r) * ldA + kb + c0;
      v4f f0 = *(const v4f*)src;
      v4f f1 = *(const v4f*)(src + 4);
      v8bf p;
      #pragma unroll
      for (int i = 0; i < 8; ++i)
        p[i] = us2bf(f2bf(i < 4 ? f0[i] : f1[i - 4]));
      *(v8bf*)&As[r][c0] = p;
    }
    // ---- B tile 32x64 stored transposed [col][k]; coalesced across columns
    #pragma unroll
    for (int s = 0; s < 2; ++s) {
      int task = tid + s * 128;          // 0..255
      int c  = task & 63;                // column within tile
      int r0 = (task >> 6) * 8;          // k chunk: 0,8,16,24
      const float* src = Bp + (long)(kb + r0) * ldB + colBase + c;
      v8bf p;
      #pragma unroll
      for (int j = 0; j < 8; ++j)
        p[j] = us2bf(f2bf(src[(long)j * ldB]));
      *(v8bf*)&Bst[c][r0] = p;
    }
    __syncthreads();

    if (kb + 32 < K) {  // hint next tiles into cache (global_prefetch_b8)
      __builtin_prefetch(&Ap[(long)(rowBase + (tid >> 1)) * ldA + kb + 32], 0, 1);
      __builtin_prefetch(&Bp[(long)(kb + 32 + (tid >> 2)) * ldB + colBase], 0, 1);
    }

    // ---- A fragment: two contiguous 16B runs -> ds_load_b128 x2
    v8bf alo = *(const v8bf*)&As[am][klo];
    v8bf ahi = *(const v8bf*)&As[am][khi];
    v16bf a = CAT16(alo, ahi);

    // ---- 4 B fragments along N (A reused)
    #pragma unroll
    for (int t = 0; t < 4; ++t) {
      const int bn = t * 16 + l16;
      v8bf blo = *(const v8bf*)&Bst[bn][klo];
      v8bf bhi = *(const v8bf*)&Bst[bn][khi];
      v16bf b = CAT16(blo, bhi);
      acc[t] = __builtin_amdgcn_wmma_f32_16x16x32_bf16(
          false, a, false, b, (short)0, acc[t], false, false);
    }
    __syncthreads();
  }

  // write back per ISA C/D layout: lanes 0-15 -> M=r, lanes 16-31 -> M=r+8
  #pragma unroll
  for (int t = 0; t < 4; ++t) {
    int col = colBase + t * 16 + l16;
    float bv = bias ? bias[col] : 0.0f;
    #pragma unroll
    for (int r = 0; r < 8; ++r) {
      int row = rowBase + wave * 16 + half * 8 + r;
      Cp[(long)row * ldC + col] = acc[t][r] + bv;
    }
  }
}

// ---------------- helper kernels ----------------
__global__ void zero_f32(float* __restrict__ p, long n) {
  long i = (long)blockIdx.x * blockDim.x + threadIdx.x;
  if (i < n) p[i] = 0.0f;
}

// deposit = v * ||k||/8, bilinear scatter to field[b,h,g,d]
__global__ void __launch_bounds__(64)
deposit_scatter(const float* __restrict__ kv, float* __restrict__ field)
{
  const int bid = blockIdx.x;        // b*N*H + n*H + h  (h in low bits)
  const int h = bid & 15;
  const int n = (bid >> 4) & 4095;
  const int b = bid >> 16;
  const int d = threadIdx.x;

  const long base = (long)(b * NTOK + n) * 2048;
  const float kval = kv[base + h * DHC + d];
  const float vval = kv[base + 1024 + h * DHC + d];

  __shared__ float red[64];
  red[d] = kval * kval;
  __syncthreads();
  #pragma unroll
  for (int s = 32; s > 0; s >>= 1) {
    if (d < s) red[d] += red[d + s];
    __syncthreads();
  }
  const float kmag = sqrtf(red[0]) / SCALE_F;
  const float dep  = vval * kmag;

  float pos = fminf((float)n * STRIDE_F, 510.0f);
  int il = (int)pos; il = il < 0 ? 0 : (il > 510 ? 510 : il);
  float frac = fminf(fmaxf(pos - (float)il, 0.0f), 1.0f);

  const long fb = (((long)(b * HC + h)) * GC + il) * DHC + d;
  atomicAdd(&field[fb],       (1.0f - frac) * dep);
  atomicAdd(&field[fb + DHC], frac * dep);
}

// T_h[g,j] = j<=g ? exp(damp*t)*cos(2*pi*f*t + phase), t=(g-j)/G
__global__ void fill_T(const float* __restrict__ freq, const float* __restrict__ damp,
                       const float* __restrict__ phase, float* __restrict__ T)
{
  long idx = (long)blockIdx.x * blockDim.x + threadIdx.x;
  if (idx >= (long)HC * GC * GC) return;
  int j = idx & (GC - 1);
  int g = (idx >> 9) & (GC - 1);
  int h = (int)(idx >> 18);
  float val = 0.0f;
  if (j <= g) {
    float t = (float)(g - j) / (float)GC;
    val = expf(damp[h] * t) * cosf(TWO_PI_F * freq[h] * t + phase[h]);
  }
  T[idx] = val;
}

// coupled[b,h,g,d] = sum_k FC[h,k] * conv[b,k,g,d]
__global__ void couple(const float* __restrict__ conv, const float* __restrict__ FC,
                       float* __restrict__ coupled)
{
  long idx = (long)blockIdx.x * blockDim.x + threadIdx.x;
  if (idx >= (long)BC * HC * GC * DHC) return;
  int d = idx & 63;
  int g = (int)(idx >> 6) & 511;
  int h = (int)(idx >> 15) & 15;
  int b = (int)(idx >> 19);
  float s = 0.0f;
  #pragma unroll
  for (int k = 0; k < HC; ++k)
    s += FC[h * HC + k] * conv[(((long)(b * HC + k)) * GC + g) * DHC + d];
  coupled[idx] = s;
}

// gathered = bilinear gather; gated = gathered * sigmoid(logits)
__global__ void gather_gate(const float* __restrict__ coupled,
                            const float* __restrict__ logits,
                            float* __restrict__ gated)
{
  long idx = (long)blockIdx.x * blockDim.x + threadIdx.x;
  if (idx >= (long)MTOK * DIMC) return;
  int c   = idx & 1023;
  long tok = idx >> 10;
  int h = c >> 6, d = c & 63;
  int n = (int)(tok & 4095), b = (int)(tok >> 12);

  float pos = fminf((float)n * STRIDE_F, 510.0f);
  int il = (int)pos; il = il < 0 ? 0 : (il > 510 ? 510 : il);
  float frac = fminf(fmaxf(pos - (float)il, 0.0f), 1.0f);

  const long cb = (((long)(b * HC + h)) * GC + il) * DHC + d;
  float merged = (1.0f - frac) * coupled[cb] + frac * coupled[cb + DHC];
  float z = logits[idx];
  float gate = 1.0f / (1.0f + expf(-z));
  gated[idx] = merged * gate;
}

// ---------------- launch ----------------
extern "C" void kernel_launch(void* const* d_in, const int* in_sizes, int n_in,
                              void* d_out, int out_size, void* d_ws, size_t ws_size,
                              hipStream_t stream)
{
  (void)in_sizes; (void)n_in; (void)out_size; (void)ws_size;
  const float* x      = (const float*)d_in[0];
  const float* qkv_w  = (const float*)d_in[1];
  const float* qkv_b  = (const float*)d_in[2];
  const float* gate_w = (const float*)d_in[3];
  const float* gate_b = (const float*)d_in[4];
  const float* out_w  = (const float*)d_in[5];
  const float* out_b  = (const float*)d_in[6];
  const float* wfreq  = (const float*)d_in[7];
  const float* wdamp  = (const float*)d_in[8];
  const float* wphase = (const float*)d_in[9];
  const float* fcoup  = (const float*)d_in[10];
  float* out = (float*)d_out;

  float* ws = (float*)d_ws;
  const long KV    = (long)MTOK * 2048;          // 33.5M floats
  const long FIELD = (long)BC * HC * GC * DHC;   // 2.1M floats
  float* kv     = ws;                // dead after scatter; reused below
  float* field  = ws + KV;
  float* conv   = field + FIELD;
  float* Tm     = conv + FIELD;      // H*G*G = 4.2M floats
  float* logits = ws;                // reuse kv region
  float* gated  = ws + (long)MTOK * DIMC;

  dim3 blk(128);

  // 1) kv = x @ qkv_w[:,1024:3072] + qkv_b[1024:]   (q is unused in reference)
  gemm_bf16_wmma<<<dim3(2048 / 64, MTOK / 64, 1), blk, 0, stream>>>(
      x, qkv_w + 1024, qkv_b + 1024, kv,
      MTOK, 2048, DIMC, DIMC, 3 * DIMC, 2048, 0, 1, 0, 0);

  // 2) zero field, 3) deposit+scatter
  zero_f32<<<(unsigned)((FIELD + 255) / 256), 256, 0, stream>>>(field, FIELD);
  deposit_scatter<<<BC * NTOK * HC, 64, 0, stream>>>(kv, field);

  // 4) per-head causal Toeplitz matrices
  {
    long n = (long)HC * GC * GC;
    fill_T<<<(unsigned)((n + 255) / 256), 256, 0, stream>>>(wfreq, wdamp, wphase, Tm);
  }

  // 5) conv[b,h] = T_h @ field[b,h]   (batched GEMM over B*H)
  gemm_bf16_wmma<<<dim3(DHC / 64, GC / 64, BC * HC), blk, 0, stream>>>(
      Tm, field, nullptr, conv,
      GC, DHC, GC, GC, DHC, DHC, (long)GC * GC, HC, (long)GC * DHC, (long)GC * DHC);

  // 6) head coupling (coupled reuses field buffer)
  float* coupled = field;
  couple<<<(unsigned)((FIELD + 255) / 256), 256, 0, stream>>>(conv, fcoup, coupled);

  // 7) gate logits = x @ gate_w + gate_b  (reuses kv region)
  gemm_bf16_wmma<<<dim3(DIMC / 64, MTOK / 64, 1), blk, 0, stream>>>(
      x, gate_w, gate_b, logits,
      MTOK, DIMC, DIMC, DIMC, DIMC, DIMC, 0, 1, 0, 0);

  // 8) gather back to tokens, apply sigmoid gate
  {
    long n = (long)MTOK * DIMC;
    gather_gate<<<(unsigned)((n + 255) / 256), 256, 0, stream>>>(coupled, logits, gated);
  }

  // 9) out = gated @ out_w + out_b
  gemm_bf16_wmma<<<dim3(DIMC / 64, MTOK / 64, 1), blk, 0, stream>>>(
      gated, out_w, out_b, out,
      MTOK, DIMC, DIMC, DIMC, DIMC, DIMC, 0, 1, 0, 0);
}
